// PredictModel_53163105190377
// MI455X (gfx1250) — matrix operand order repaired and checked by reading, not compiled, not run on hardware
//
#include <hip/hip_runtime.h>
#include <hip/hip_bf16.h>

// Problem constants (from reference module: B,S,D,H,L, N_HIST, M_FUT)
#define B_  512
#define S_  256
#define D_  32
#define H_  256
#define L_  128
#define T_  80          // N_HIST + M_FUT

#define THREADS_ 512    // 16 wave32s per block

typedef __attribute__((ext_vector_type(16))) __bf16 v16bf;
typedef __attribute__((ext_vector_type(8)))  __bf16 v8bf;
typedef __attribute__((ext_vector_type(8)))  float  v8f;

__device__ __forceinline__ v8f wmma_bf16(v16bf a, v16bf b, v8f c) {
  // D = A(16x32 bf16) * B(32x16 bf16) + C(16x16 f32)
  return __builtin_amdgcn_wmma_f32_16x16x32_bf16(false, a, false, b, (short)0, c,
                                                 false, false);
}

// A fragment (16x32 bf16) from a row-major BF16 tile [16][ldk] in LDS, K-block kblk.
// Lanes 0-15: row m=lane, K {kblk..+7, kblk+16..+23}; lanes 16-31: K offset +8.
// Two aligned 16B vector loads, no conversion VALU -> no WMMA WAR hazard NOPs.
__device__ __forceinline__ v16bf load_a_bf16(const __bf16* src, int ldk, int kblk, int lane) {
  const int m  = lane & 15;
  const int ko = (lane >> 4) << 3;
  const __bf16* p = src + (size_t)m * ldk + kblk + ko;
  const v8bf lo = *(const v8bf*)p;
  const v8bf hi = *(const v8bf*)(p + 16);
  return __builtin_shufflevector(lo, hi, 0, 1, 2, 3, 4, 5, 6, 7,
                                         8, 9, 10, 11, 12, 13, 14, 15);
}

// A fragment from a row-major FP32 source (used only for global x_t).
__device__ __forceinline__ v16bf load_a_f32(const float* src, int ldk, int kblk, int lane) {
  const int m  = lane & 15;
  const int ko = (lane >> 4) << 3;
  const float* p = src + (size_t)m * ldk + kblk + ko;
  v16bf a;
#pragma unroll
  for (int i = 0; i < 8; ++i) a[i] = (__bf16)p[i];
#pragma unroll
  for (int i = 0; i < 8; ++i) a[8 + i] = (__bf16)p[16 + i];
  return a;
}

// B fragment (32x16 bf16) = tile of W^T, W row-major [N][ldk] in bf16.
// B[k][n] = W[n][k]; per lane: K {kblk+ko..+7, kblk+ko+16..+23}, ko = 0/8 -> two aligned 16B loads.
__device__ __forceinline__ v16bf load_b_w(const __bf16* W, int ldk, int ntile, int kblk, int lane) {
  const int n  = lane & 15;
  const int ko = (lane < 16) ? 0 : 8;
  const __bf16* p = W + (size_t)(ntile * 16 + n) * ldk + kblk + ko;
  const v8bf lo = *(const v8bf*)p;
  const v8bf hi = *(const v8bf*)(p + 16);
  return __builtin_shufflevector(lo, hi, 0, 1, 2, 3, 4, 5, 6, 7,
                                         8, 9, 10, 11, 12, 13, 14, 15);
}

__device__ __forceinline__ float softplus_f(float v) {
  return (v > 15.f) ? v : __logf(1.f + __expf(v));
}
__device__ __forceinline__ float sigmoid_f(float v) {
  return 1.f / (1.f + __expf(-v));
}

// ODE func: dst = softplus(src @ w1^T + b1) @ w2^T + b2   ([16][128] per block)
// srcb/s1b are bf16 GEMM-input buffers; dst is fp32. Waves 0..7 own one tile each.
__device__ __forceinline__ void ode_f(const __bf16* srcb, float* dst, __bf16* s1b,
                                      const __bf16* w1, const __bf16* w2,
                                      const float* b1, const float* b2,
                                      int wv, int lane, int mo, int n15) {
  if (wv < 8) {
    v8f acc = {};
#pragma unroll
    for (int kb = 0; kb < 4; ++kb)
      acc = wmma_bf16(load_a_bf16(srcb, L_, kb * 32, lane),
                      load_b_w(w1, L_, wv, kb * 32, lane), acc);
    const int c = wv * 16 + n15;
#pragma unroll
    for (int r = 0; r < 8; ++r)
      s1b[(r + mo) * L_ + c] = (__bf16)softplus_f(acc[r] + b1[c]);
  }
  __syncthreads();
  if (wv < 8) {
    v8f acc = {};
#pragma unroll
    for (int kb = 0; kb < 4; ++kb)
      acc = wmma_bf16(load_a_bf16(s1b, L_, kb * 32, lane),
                      load_b_w(w2, L_, wv, kb * 32, lane), acc);
    const int c = wv * 16 + n15;
#pragma unroll
    for (int r = 0; r < 8; ++r)
      dst[(r + mo) * L_ + c] = acc[r] + b2[c];
  }
  __syncthreads();
}

// out[b, t, :] = y @ w_dec^T + b_dec   (D_=32 -> 2 N-tiles, waves 0 and 1)
__device__ __forceinline__ void decode_t(const __bf16* yb, int t, float* out,
                                         const __bf16* wdec, const float* b_dec,
                                         int rowbase, int wv, int lane, int mo, int n15) {
  if (wv < 2) {
    v8f acc = {};
#pragma unroll
    for (int kb = 0; kb < 4; ++kb)
      acc = wmma_bf16(load_a_bf16(yb, L_, kb * 32, lane),
                      load_b_w(wdec, L_, wv, kb * 32, lane), acc);
    const int d = wv * 16 + n15;
#pragma unroll
    for (int r = 0; r < 8; ++r) {
      const int b = rowbase + r + mo;
      out[((size_t)b * T_ + t) * D_ + d] = acc[r] + b_dec[d];
    }
  }
}

__global__ void cvt_bf16_kernel(const float* __restrict__ s, __bf16* __restrict__ d, int n) {
  int i = blockIdx.x * blockDim.x + threadIdx.x;
  if (i < n) d[i] = (__bf16)s[i];
}

// One block = 16 batch rows, 16 wave32s. Entire model per block, no global sync.
__launch_bounds__(THREADS_)
__global__ void fused_latent_ode(
    const float* __restrict__ x,    const float* __restrict__ eps,
    const float* __restrict__ b_ih, const float* __restrict__ b_hh,
    const float* __restrict__ b_mu, const float* __restrict__ b_lv,
    const float* __restrict__ b_g,  const float* __restrict__ b1,
    const float* __restrict__ b2,   const float* __restrict__ b_dec,
    const __bf16* __restrict__ whh, const __bf16* __restrict__ wih,
    const __bf16* __restrict__ wmu, const __bf16* __restrict__ wlv,
    const __bf16* __restrict__ wg,  const __bf16* __restrict__ w1,
    const __bf16* __restrict__ w2,  const __bf16* __restrict__ wdec,
    float* __restrict__ out, float* __restrict__ mu_out, float* __restrict__ lv_out)
{
  // 44 KB LDS
  __shared__ alignas(16) unsigned char smem[45056];
  __bf16* h16   = (__bf16*)(smem);          // [16][256] bf16: h (later z0) GEMM input
  float*  y     = (float*)(smem + 8192);    // [16][128] fp32 ODE state master
  float*  ka    = (float*)(smem + 16384);   // [16][128] fp32 current k_i
  float*  ksum  = (float*)(smem + 24576);   // [16][128] fp32 running RK4 sum
  __bf16* yb    = (__bf16*)(smem + 32768);  // [16][128] bf16 shadow of y
  __bf16* ytmpb = (__bf16*)(smem + 36864);  // [16][128] bf16 RK4 midpoint state
  __bf16* s1b   = (__bf16*)(smem + 40960);  // [16][128] bf16 softplus intermediate

  const int tid  = threadIdx.x;
  const int lane = tid & 31;
  const int wv   = tid >> 5;               // wave 0..15
  const int rowbase = blockIdx.x * 16;
  const int mo  = (lane >> 4) << 3;        // C/D row offset for this lane half
  const int n15 = lane & 15;

  for (int i = tid; i < 16 * H_; i += THREADS_) h16[i] = (__bf16)0.f;

  // GRU: wave wv owns column block [16wv, 16wv+16) of H for each gate
  // (tiles: r -> wv, z -> 16+wv, n -> 32+wv), so gates stay register-aligned.
  const int c0 = 16 * wv + n15;
  const float bihR = b_ih[c0],       bhhR = b_hh[c0];
  const float bihZ = b_ih[256 + c0], bhhZ = b_hh[256 + c0];
  const float bihN = b_ih[512 + c0], bhhN = b_hh[512 + c0];

  // t-invariant w_ih B-fragments (K=32 exactly): keep resident in registers
  const v16bf wihR = load_b_w(wih, D_,  0 + wv, 0, lane);
  const v16bf wihZ = load_b_w(wih, D_, 16 + wv, 0, lane);
  const v16bf wihN = load_b_w(wih, D_, 32 + wv, 0, lane);

  // fp32 master of h for this lane's owned (row, col) elements lives in registers:
  // element r corresponds to h[row = r + mo][col = c0].
  float hreg[8];
#pragma unroll
  for (int r = 0; r < 8; ++r) hreg[r] = 0.f;

  __syncthreads();

  // ---------------- GRU over S_ steps ----------------
  for (int t = 0; t < S_; ++t) {
    // x_t contribution initializes the r/z accumulators (x-part and h-part of
    // r,z are summed pre-sigmoid anyway); only the n gate needs x/h separate.
    const v16bf axf = load_a_f32(x + ((size_t)rowbase * S_ + t) * D_, S_ * D_, 0, lane);
    v8f zero = {};
    v8f accR  = wmma_bf16(axf, wihR, zero);
    v8f accZ  = wmma_bf16(axf, wihZ, zero);
    v8f accXN = wmma_bf16(axf, wihN, zero);
    v8f accHN = {};

    // h @ w_hh^T; A fragments are raw bf16 LDS loads (no cvt -> no WAR NOPs),
    // double-buffered across K-steps.
    v16bf afA = load_a_bf16(h16, H_, 0, lane);
#pragma unroll 2
    for (int kb = 0; kb < 8; ++kb) {
      const v16bf bR = load_b_w(whh, H_,  0 + wv, kb * 32, lane);
      const v16bf bZ = load_b_w(whh, H_, 16 + wv, kb * 32, lane);
      const v16bf bN = load_b_w(whh, H_, 32 + wv, kb * 32, lane);
      const v16bf afN = load_a_bf16(h16, H_, ((kb + 1) & 7) * 32, lane);
      accR  = wmma_bf16(afA, bR, accR);
      accZ  = wmma_bf16(afA, bZ, accZ);
      accHN = wmma_bf16(afA, bN, accHN);
      afA = afN;
    }

    // Gate combine entirely in registers; h_old is this wave's own hreg
#pragma unroll
    for (int r = 0; r < 8; ++r) {
      const float rg = sigmoid_f(accR[r] + bihR + bhhR);
      const float zg = sigmoid_f(accZ[r] + bihZ + bhhZ);
      const float cd = tanhf(accXN[r] + bihN + rg * (accHN[r] + bhhN));
      hreg[r] = (1.f - zg) * cd + zg * hreg[r];
    }
    __syncthreads();                       // all A-fragment reads of h16 done
#pragma unroll
    for (int r = 0; r < 8; ++r)
      h16[(r + mo) * H_ + c0] = (__bf16)hreg[r];
    __syncthreads();
  }

  // ---------------- VAE head: mu, logvar, z0 = mu + exp(0.5 lv)*eps ----------------
  {
    // 16 N-tiles each for mu and logvar -> wave wv owns tile wv of both
    v8f am = {};
    v8f al = {};
    v16bf afA = load_a_bf16(h16, H_, 0, lane);
#pragma unroll 2
    for (int kb = 0; kb < 8; ++kb) {
      const v16bf bm = load_b_w(wmu, H_, wv, kb * 32, lane);
      const v16bf bl = load_b_w(wlv, H_, wv, kb * 32, lane);
      const v16bf afN = load_a_bf16(h16, H_, ((kb + 1) & 7) * 32, lane);
      am = wmma_bf16(afA, bm, am);
      al = wmma_bf16(afA, bl, al);
      afA = afN;
    }
    float z0v[8];
#pragma unroll
    for (int r = 0; r < 8; ++r) {
      const int b = rowbase + r + mo;
      const float mv = am[r] + b_mu[c0];
      const float lv = al[r] + b_lv[c0];
      mu_out[(size_t)b * H_ + c0] = mv;
      lv_out[(size_t)b * H_ + c0] = lv;
      z0v[r] = mv + __expf(0.5f * lv) * eps[(size_t)b * H_ + c0];
    }
    __syncthreads();                       // h no longer needed; reuse h16 for z0
#pragma unroll
    for (int r = 0; r < 8; ++r)
      h16[(r + mo) * H_ + c0] = (__bf16)z0v[r];
    __syncthreads();
  }

  // ---------------- y0 = z0 @ w_g^T + b_g  ([16][128], waves 0..7) ----------------
  if (wv < 8) {
    v8f acc = {};
#pragma unroll
    for (int kb = 0; kb < 8; ++kb)
      acc = wmma_bf16(load_a_bf16(h16, H_, kb * 32, lane),
                      load_b_w(wg, H_, wv, kb * 32, lane), acc);
    const int c = wv * 16 + n15;
#pragma unroll
    for (int r = 0; r < 8; ++r) {
      const float v = acc[r] + b_g[c];
      y[(r + mo) * L_ + c]  = v;
      yb[(r + mo) * L_ + c] = (__bf16)v;
    }
  }
  __syncthreads();

  decode_t(yb, 0, out, wdec, b_dec, rowbase, wv, lane, mo, n15);

  // ---------------- RK4 integration, 79 steps, decode each ----------------
  for (int st = 0; st < T_ - 1; ++st) {
    ode_f(yb, ka, s1b, w1, w2, b1, b2, wv, lane, mo, n15);          // k1
    for (int e = tid; e < 16 * L_; e += THREADS_) {
      const float k = ka[e]; ksum[e] = k; ytmpb[e] = (__bf16)(y[e] + 0.5f * k);
    }
    __syncthreads();
    ode_f(ytmpb, ka, s1b, w1, w2, b1, b2, wv, lane, mo, n15);       // k2
    for (int e = tid; e < 16 * L_; e += THREADS_) {
      const float k = ka[e]; ksum[e] += 2.f * k; ytmpb[e] = (__bf16)(y[e] + 0.5f * k);
    }
    __syncthreads();
    ode_f(ytmpb, ka, s1b, w1, w2, b1, b2, wv, lane, mo, n15);       // k3
    for (int e = tid; e < 16 * L_; e += THREADS_) {
      const float k = ka[e]; ksum[e] += 2.f * k; ytmpb[e] = (__bf16)(y[e] + k);
    }
    __syncthreads();
    ode_f(ytmpb, ka, s1b, w1, w2, b1, b2, wv, lane, mo, n15);       // k4
    for (int e = tid; e < 16 * L_; e += THREADS_) {
      const float v = y[e] + (1.f / 6.f) * (ksum[e] + ka[e]);
      y[e]  = v;
      yb[e] = (__bf16)v;
    }
    __syncthreads();
    decode_t(yb, st + 1, out, wdec, b_dec, rowbase, wv, lane, mo, n15);
  }
}

extern "C" void kernel_launch(void* const* d_in, const int* in_sizes, int n_in,
                              void* d_out, int out_size, void* d_ws, size_t ws_size,
                              hipStream_t stream) {
  (void)in_sizes; (void)n_in; (void)out_size; (void)ws_size;
  const float* x     = (const float*)d_in[0];
  const float* eps   = (const float*)d_in[1];
  const float* w_ih  = (const float*)d_in[2];
  const float* w_hh  = (const float*)d_in[3];
  const float* b_ih  = (const float*)d_in[4];
  const float* b_hh  = (const float*)d_in[5];
  const float* w_mu  = (const float*)d_in[6];
  const float* b_mu  = (const float*)d_in[7];
  const float* w_lv  = (const float*)d_in[8];
  const float* b_lv  = (const float*)d_in[9];
  const float* w_g   = (const float*)d_in[10];
  const float* b_g   = (const float*)d_in[11];
  const float* w1    = (const float*)d_in[12];
  const float* b1    = (const float*)d_in[13];
  const float* w2    = (const float*)d_in[14];
  const float* b2    = (const float*)d_in[15];
  const float* w_dec = (const float*)d_in[16];
  const float* b_dec = (const float*)d_in[17];

  // Workspace: bf16 copies of all weight matrices (~0.85 MB total, L2-resident)
  __bf16* whh = (__bf16*)d_ws;             // 768*256
  __bf16* wih = whh + 768 * 256;           // 768*32
  __bf16* wmu = wih + 768 * 32;            // 256*256
  __bf16* wlv = wmu + 256 * 256;           // 256*256
  __bf16* wgb = wlv + 256 * 256;           // 128*256
  __bf16* w1b = wgb + 128 * 256;           // 128*128
  __bf16* w2b = w1b + 128 * 128;           // 128*128
  __bf16* wdb = w2b + 128 * 128;           // 32*128

  auto cvt = [&](const float* s, __bf16* d, int n) {
    cvt_bf16_kernel<<<(n + 255) / 256, 256, 0, stream>>>(s, d, n);
  };
  cvt(w_hh, whh, 768 * 256);
  cvt(w_ih, wih, 768 * 32);
  cvt(w_mu, wmu, 256 * 256);
  cvt(w_lv, wlv, 256 * 256);
  cvt(w_g,  wgb, 128 * 256);
  cvt(w1,   w1b, 128 * 128);
  cvt(w2,   w2b, 128 * 128);
  cvt(w_dec, wdb, 32 * 128);

  float* outp   = (float*)d_out;                       // [B, T, D]
  float* mu_out = outp + (size_t)B_ * T_ * D_;         // [B, H]
  float* lv_out = mu_out + (size_t)B_ * H_;            // [B, H]

  fused_latent_ode<<<B_ / 16, THREADS_, 0, stream>>>(
      x, eps, b_ih, b_hh, b_mu, b_lv, b_g, b1, b2, b_dec,
      whh, wih, wmu, wlv, wgb, w1b, w2b, wdb,
      outp, mu_out, lv_out);
}
